// Gradformer_DownstreamTasks_8976481649039
// MI455X (gfx1250) — matrix operand order repaired and checked by reading, not compiled, not run on hardware
//
#include <hip/hip_runtime.h>
#include <stdint.h>

// ---------------------------------------------------------------------------
// Gradformer forward for MI455X (gfx1250): all dense GEMMs + attention on
// v_wmma_f32_16x16x32_bf16 (wave32, 16x16x32 tiles, f32 accumulate).
// GEMM: one wave = 16x64 D tile (4 WMMAs/k-step, A fragment reused).
// ---------------------------------------------------------------------------

typedef __bf16 bf16_t;
typedef bf16_t v16bf __attribute__((ext_vector_type(16)));
typedef float  v8f   __attribute__((ext_vector_type(8)));
typedef unsigned short u16;
typedef unsigned int   u32;

#define NT    16384   // B*N nodes
#define Bg    64
#define Nn    256
#define Ch    128     // channels C
#define Hh    4       // heads
#define DHd   32      // C/H
#define Ll    10      // layers
#define Ee    131072  // edges
#define EMBA  108     // C - PE_DIM
#define PED   20

__device__ __forceinline__ u16 f2bf(float f) {
  u32 u = __float_as_uint(f);
  return (u16)((u + 0x7FFFu + ((u >> 16) & 1u)) >> 16);  // RNE
}

union FragU {
  uint4 q[2];
  u16   u[16];
  v16bf v;
};

// A-matrix 16x32 bf16 fragment (row-major source, leading dim ldk):
// lane L: M = L&15 ; K window base = (L>=16 ? 8 : 0); elems 0-7 -> K base+0..7,
// elems 8-15 -> K base+16..23  (ISA 7.12.2, 16-bit A 16x32 layout)
__device__ __forceinline__ v16bf load_a_frag(const u16* __restrict__ X, int ldk,
                                             int row, int k0, int lane) {
  const u16* p = X + (size_t)row * ldk + k0 + ((lane >= 16) ? 8 : 0);
  FragU t;
  t.q[0] = *(const uint4*)(p);
  t.q[1] = *(const uint4*)(p + 16);
  return t.v;
}

// B-matrix 32x16 bf16 fragment from K-contiguous storage (W^T, row-major
// [N, K]): lane L: Ncol = L&15 ; K base = (L>=16 ? 8 : 0); elems 0-7 -> K base+0..7, elems 8-15 -> K base+16..23
__device__ __forceinline__ v16bf load_b_frag(const u16* __restrict__ Wt, int ldk,
                                             int col, int k0, int lane) {
  const u16* p = Wt + (size_t)col * ldk + k0 + ((lane >= 16) ? 8 : 0);
  FragU t;
  t.q[0] = *(const uint4*)(p);
  t.q[1] = *(const uint4*)(p + 16);
  return t.v;
}

__device__ __forceinline__ v8f wmma_bf16(v16bf a, v16bf b, v8f c) {
  return __builtin_amdgcn_wmma_f32_16x16x32_bf16(false, a, false, b, (short)0, c,
                                                 false, false);
}

// ---------------------------------------------------------------------------
// Encoders
// ---------------------------------------------------------------------------
struct EncArgs { const float* at[9]; const float* pw; const float* pb; };
struct BondArgs { const float* bt[3]; };

__global__ __launch_bounds__(Ch) void encode_kernel(EncArgs ep, const int* __restrict__ x,
                                                    const float* __restrict__ pe,
                                                    float* __restrict__ h,
                                                    u16* __restrict__ hb) {
  int n = blockIdx.x, c = threadIdx.x;
  float v;
  if (c < EMBA) {
    v = 0.f;
#pragma unroll
    for (int i = 0; i < 9; ++i) v += ep.at[i][(size_t)x[n * 9 + i] * EMBA + c];
  } else {
    int c2 = c - EMBA;
    v = ep.pb[c2];
#pragma unroll
    for (int k = 0; k < PED; ++k) v += pe[n * PED + k] * ep.pw[k * PED + c2];
  }
  h[(size_t)n * Ch + c]  = v;
  hb[(size_t)n * Ch + c] = f2bf(v);
}

__global__ __launch_bounds__(Ch) void bond_kernel(BondArgs bp, const int* __restrict__ ea,
                                                  float* __restrict__ e) {
  int j = blockIdx.x, c = threadIdx.x;
  float v = 0.f;
#pragma unroll
  for (int i = 0; i < 3; ++i) v += bp.bt[i][(size_t)ea[j * 3 + i] * Ch + c];
  e[(size_t)j * Ch + c] = v;
}

// ---------------------------------------------------------------------------
// GINE message passing: agg[dst] += relu(h[src] + e)
// ---------------------------------------------------------------------------
__global__ __launch_bounds__(256) void zero_f32(float* p, int n) {
  int i = blockIdx.x * 256 + threadIdx.x;
  if (i < n) p[i] = 0.f;
}

__global__ __launch_bounds__(256) void gine_msg(const float* __restrict__ h,
                                                const float* __restrict__ e,
                                                const int* __restrict__ src,
                                                const int* __restrict__ dst,
                                                float* __restrict__ agg) {
  int idx = blockIdx.x * 256 + threadIdx.x;   // Ee*Ch threads
  int ed = idx >> 7, c = idx & (Ch - 1);
  float m = h[(size_t)src[ed] * Ch + c] + e[idx];
  if (m > 0.f) atomicAdd(&agg[(size_t)dst[ed] * Ch + c], m);
}

__global__ __launch_bounds__(256) void gine_combine(const float* __restrict__ h,
                                                    const float* __restrict__ agg,
                                                    const float* __restrict__ eps,
                                                    u16* __restrict__ xb) {
  int i = blockIdx.x * 256 + threadIdx.x;
  xb[i] = f2bf((1.f + eps[0]) * h[i] + agg[i]);
}

// ---------------------------------------------------------------------------
// Generic WMMA GEMM: Y = act(Xb[M,K] @ W[K,Nc] + bias) (+ res), W given as
// bf16 W^T [Nc,K]. One wave computes a 16x64 D tile; A fragment loaded once
// per 32-wide k-step and reused by 4 WMMAs (4x arithmetic intensity).
// ---------------------------------------------------------------------------
template <int RELU>
__global__ __launch_bounds__(256) void gemm_k(const u16* __restrict__ Xb,
                                              const u16* __restrict__ Wt,
                                              const float* __restrict__ bias,
                                              const float* __restrict__ res,
                                              float* __restrict__ Yf,
                                              u16* __restrict__ Yb,
                                              int M, int K, int Nc) {
  int gw   = blockIdx.x * 8 + (threadIdx.x >> 5);
  int lane = threadIdx.x & 31;
  int ntil = Nc >> 6;                    // 64-column groups
  int tM = gw / ntil, tN = gw - tM * ntil;
  if (tM >= (M >> 4)) return;            // uniform per wave (EXEC stays all-1)

  int arow  = tM * 16 + (lane & 15);
  int bcol0 = tN * 64 + (lane & 15);
  v8f acc[4];
#pragma unroll
  for (int c = 0; c < 4; ++c) acc[c] = (v8f){0.f, 0.f, 0.f, 0.f, 0.f, 0.f, 0.f, 0.f};

  for (int k0 = 0; k0 < K; k0 += 32) {
    // prefetch next k-step (lowers to global_prefetch_b8)
    if (k0 + 32 < K) {
      __builtin_prefetch(Xb + (size_t)arow * K + k0 + 32, 0, 1);
      __builtin_prefetch(Wt + (size_t)bcol0 * K + k0 + 32, 0, 1);
    }
    v16bf a = load_a_frag(Xb, K, arow, k0, lane);
#pragma unroll
    for (int c = 0; c < 4; ++c) {
      v16bf b = load_b_frag(Wt, K, bcol0 + c * 16, k0, lane);
      acc[c] = wmma_bf16(a, b, acc[c]);
    }
  }

#pragma unroll
  for (int c = 0; c < 4; ++c) {
    int bcol = bcol0 + c * 16;
    float bc = bias ? bias[bcol] : 0.f;
#pragma unroll
    for (int r = 0; r < 8; ++r) {
      int orow = tM * 16 + r + ((lane >= 16) ? 8 : 0);  // C/D layout (ISA 7.12.2)
      size_t o = (size_t)orow * Nc + bcol;
      float v = acc[c][r] + bc;
      if (RELU) v = fmaxf(v, 0.f);
      if (res) v += res[o];
      if (Yf) Yf[o] = v;
      if (Yb) Yb[o] = f2bf(v);
    }
  }
}

// ---------------------------------------------------------------------------
// Fused hop-decayed attention. One wave per (b, head, 16-row tile).
// scores = (Q K^T) * scale * 0.5^max(sph - hop, 0); softmax; O = P V.
// ---------------------------------------------------------------------------
__global__ __launch_bounds__(32) void attn_kernel(const u16* __restrict__ qb,
                                                  const u16* __restrict__ kb,
                                                  const u16* __restrict__ vb,
                                                  const float* __restrict__ sph,
                                                  const float* __restrict__ hop,
                                                  u16* __restrict__ ob) {
  __shared__ u16  Ks[Nn * DHd];   // 16 KB
  __shared__ u16  Vs[Nn * DHd];   // 16 KB
  __shared__ float Sc[16 * Nn];   // 16 KB
  __shared__ u16  Ps[16 * Nn];    //  8 KB
  int blk = blockIdx.x;                 // Bg*Hh*16
  int it = blk & 15, hh = (blk >> 4) & (Hh - 1), b = blk >> 6;
  int lane = threadIdx.x;

  for (int j = lane; j < Nn; j += 32) {
    const uint4* ks = (const uint4*)(kb + ((size_t)(b * Nn + j)) * Ch + hh * DHd);
    const uint4* vs = (const uint4*)(vb + ((size_t)(b * Nn + j)) * Ch + hh * DHd);
    uint4* kd = (uint4*)(Ks + j * DHd);
    uint4* vd = (uint4*)(Vs + j * DHd);
#pragma unroll
    for (int t = 0; t < 4; ++t) { kd[t] = ks[t]; vd[t] = vs[t]; }
  }
  __syncthreads();

  // Q fragment for this row tile (K-dim = DH = 32 -> one WMMA per j-tile)
  int qrow = it * 16 + (lane & 15);
  const u16* qp = qb + ((size_t)(b * Nn + qrow)) * Ch + hh * DHd + ((lane >= 16) ? 8 : 0);
  FragU qa;
  qa.q[0] = *(const uint4*)(qp);
  qa.q[1] = *(const uint4*)(qp + 16);

  const float scale = 0.17677669529f;  // 1/sqrt(32)
  float hv = hop[hh];
  for (int jt = 0; jt < 16; ++jt) {
    const u16* kp = Ks + (jt * 16 + (lane & 15)) * DHd + ((lane >= 16) ? 8 : 0);
    FragU kf;
    kf.q[0] = *(const uint4*)(kp);
    kf.q[1] = *(const uint4*)(kp + 16);
    v8f acc = {0.f, 0.f, 0.f, 0.f, 0.f, 0.f, 0.f, 0.f};
    acc = wmma_bf16(qa.v, kf.v, acc);
    int j = jt * 16 + (lane & 15);
#pragma unroll
    for (int r = 0; r < 8; ++r) {
      int il = r + ((lane >= 16) ? 8 : 0);
      int gi = it * 16 + il;
      float d = sph[((size_t)b * Nn + gi) * Nn + j] - hv;
      float sp = exp2f(-fmaxf(d, 0.f));          // gamma=0.5, slope=0
      Sc[il * Nn + j] = acc[r] * scale * sp;
    }
  }
  __syncthreads();

  // Row-wise softmax (each lane owns 8 contiguous columns of each row)
  for (int r = 0; r < 16; ++r) {
    float ev[8], m = -3.0e38f;
#pragma unroll
    for (int t = 0; t < 8; ++t) m = fmaxf(m, Sc[r * Nn + lane * 8 + t]);
    for (int o = 16; o; o >>= 1) m = fmaxf(m, __shfl_xor(m, o, 32));
    float s = 0.f;
#pragma unroll
    for (int t = 0; t < 8; ++t) {
      ev[t] = exp2f((Sc[r * Nn + lane * 8 + t] - m) * 1.44269504f);
      s += ev[t];
    }
    for (int o = 16; o; o >>= 1) s += __shfl_xor(s, o, 32);
    float inv = 1.f / s;
#pragma unroll
    for (int t = 0; t < 8; ++t) Ps[r * Nn + lane * 8 + t] = f2bf(ev[t] * inv);
  }
  __syncthreads();

  // O tile = P[16x256] @ V[256x32] : 8 K-chunks x 2 column blocks
  v8f o0 = {0.f, 0.f, 0.f, 0.f, 0.f, 0.f, 0.f, 0.f};
  v8f o1 = {0.f, 0.f, 0.f, 0.f, 0.f, 0.f, 0.f, 0.f};
  for (int kc = 0; kc < 8; ++kc) {
    const u16* pa = Ps + (lane & 15) * Nn + kc * 32 + ((lane >= 16) ? 16 : 0);
    FragU af;
    af.q[0] = *(const uint4*)(pa);
    af.q[1] = *(const uint4*)(pa + 8);
    int kb0 = kc * 32 + ((lane >= 16) ? 16 : 0);
    FragU b0, b1;
#pragma unroll
    for (int t = 0; t < 16; ++t) {
      b0.u[t] = Vs[(kb0 + t) * DHd + (lane & 15)];
      b1.u[t] = Vs[(kb0 + t) * DHd + 16 + (lane & 15)];
    }
    o0 = wmma_bf16(af.v, b0.v, o0);
    o1 = wmma_bf16(af.v, b1.v, o1);
  }
#pragma unroll
  for (int r = 0; r < 8; ++r) {
    int gi = it * 16 + r + ((lane >= 16) ? 8 : 0);
    size_t base = ((size_t)(b * Nn + gi)) * Ch + hh * DHd;
    ob[base + (lane & 15)]      = f2bf(o0[r]);
    ob[base + 16 + (lane & 15)] = f2bf(o1[r]);
  }
}

// ---------------------------------------------------------------------------
// BatchNorm1d (training stats) per channel; elementwise helpers; pooling.
// ---------------------------------------------------------------------------
__global__ __launch_bounds__(256) void bn_kernel(const float* __restrict__ X,
                                                 const float* __restrict__ g,
                                                 const float* __restrict__ bt,
                                                 float* __restrict__ Yf,
                                                 u16* __restrict__ Yb, int M) {
  __shared__ float s0[256], s1[256];
  int c = blockIdx.x, t = threadIdx.x;
  float sum = 0.f, sq = 0.f;
  for (int r = t; r < M; r += 256) {
    float v = X[(size_t)r * Ch + c];
    sum += v; sq += v * v;
  }
  s0[t] = sum; s1[t] = sq;
  __syncthreads();
  for (int o = 128; o; o >>= 1) {
    if (t < o) { s0[t] += s0[t + o]; s1[t] += s1[t + o]; }
    __syncthreads();
  }
  float mu = s0[0] / M;
  float var = s1[0] / M - mu * mu;
  float is = rsqrtf(var + 1e-5f) * g[c];
  float bb = bt[c];
  for (int r = t; r < M; r += 256) {
    float v = (X[(size_t)r * Ch + c] - mu) * is + bb;
    if (Yf) Yf[(size_t)r * Ch + c] = v;
    if (Yb) Yb[(size_t)r * Ch + c] = f2bf(v);
  }
}

__global__ __launch_bounds__(256) void ew_add(const float* __restrict__ a,
                                              const float* __restrict__ b,
                                              float* __restrict__ yf,
                                              u16* __restrict__ yb) {
  int i = blockIdx.x * 256 + threadIdx.x;
  float v = a[i] + b[i];
  yf[i] = v;
  yb[i] = f2bf(v);
}

__global__ __launch_bounds__(256) void transpose_bf16(const float* __restrict__ W,
                                                      u16* __restrict__ Wt,
                                                      int K, int Nc) {
  int t = blockIdx.x * 256 + threadIdx.x;
  if (t >= K * Nc) return;
  int k = t / Nc, n = t - k * Nc;
  Wt[(size_t)n * K + k] = f2bf(W[t]);
}

__global__ __launch_bounds__(256) void pool_kernel(const float* __restrict__ h,
                                                   float* __restrict__ out) {
  int t = blockIdx.x * 256 + threadIdx.x;  // Bg*Ch threads
  int b = t / Ch, c = t - b * Ch;
  float s = 0.f;
  for (int i = 0; i < Nn; ++i) s += h[((size_t)b * Nn + i) * Ch + c];
  out[t] = s * (1.f / Nn);
}

// ---------------------------------------------------------------------------
// Host orchestration
// ---------------------------------------------------------------------------
enum {
  I_bk = 0, I_bn1_b, I_bn1_g, I_bn2_b, I_bn2_g, I_bn3_b, I_bn3_g, I_bo, I_bq,
  I_bv, I_eps, I_ffn_b1, I_ffn_b2, I_ffn_w1, I_ffn_w2, I_gine_b1, I_gine_b2,
  I_gine_w1, I_gine_w2, I_wk, I_wo, I_wq, I_wv
};

static const size_t LAYER_W = 163840;  // bf16 elems of transposed weights/layer

static void gemm_launch(bool relu, const u16* X, const u16* Wt, const float* bias,
                        const float* res, float* Yf, u16* Yb, int M, int K, int Nc,
                        hipStream_t s) {
  int waves = (M / 16) * (Nc / 64);      // one wave per 16x64 tile
  dim3 g(waves / 8), b(256);
  if (relu)
    gemm_k<1><<<g, b, 0, s>>>(X, Wt, bias, res, Yf, Yb, M, K, Nc);
  else
    gemm_k<0><<<g, b, 0, s>>>(X, Wt, bias, res, Yf, Yb, M, K, Nc);
}

extern "C" void kernel_launch(void* const* d_in, const int* in_sizes, int n_in,
                              void* d_out, int out_size, void* d_ws, size_t ws_size,
                              hipStream_t stream) {
  (void)in_sizes; (void)n_in; (void)out_size; (void)ws_size;
  const int*   x   = (const int*)d_in[0];
  const float* pe  = (const float*)d_in[1];
  const int*   ei  = (const int*)d_in[2];
  const int*   ea  = (const int*)d_in[3];
  const float* sph = (const float*)d_in[5];

  // params pytree leaves (dict keys sorted): atom_emb[9], bond_emb[3], hop,
  // layers[10]{23 sorted leaves}, pe_lin_b, pe_lin_w
  EncArgs ep;
  for (int i = 0; i < 9; ++i) ep.at[i] = (const float*)d_in[6 + i];
  BondArgs bp;
  for (int i = 0; i < 3; ++i) bp.bt[i] = (const float*)d_in[15 + i];
  const float* hop = (const float*)d_in[18];
  ep.pb = (const float*)d_in[19 + Ll * 23];
  ep.pw = (const float*)d_in[20 + Ll * 23];
  auto LP = [&](int l, int k) { return (const float*)d_in[19 + l * 23 + k]; };

  // ---- workspace carve-up ----
  auto alg = [](size_t v) { return (v + 255) & ~(size_t)255; };
  char* w = (char*)d_ws;
  float* h   = (float*)w; w += alg((size_t)NT * Ch * 4);
  u16*   hb  = (u16*)w;   w += alg((size_t)NT * Ch * 2);
  float* e   = (float*)w; w += alg((size_t)Ee * Ch * 4);
  float* agg = (float*)w; w += alg((size_t)NT * Ch * 4);
  float* t0  = (float*)w; w += alg((size_t)NT * Ch * 4);
  float* t1  = (float*)w; w += alg((size_t)NT * Ch * 4);
  float* t2  = (float*)w; w += alg((size_t)NT * Ch * 4);
  u16*   xb  = (u16*)w;   w += alg((size_t)NT * 2 * Ch * 2);
  u16*   xb2 = (u16*)w;   w += alg((size_t)NT * 2 * Ch * 2);
  u16*   qb  = (u16*)w;   w += alg((size_t)NT * Ch * 2);
  u16*   kb  = (u16*)w;   w += alg((size_t)NT * Ch * 2);
  u16*   vb  = (u16*)w;   w += alg((size_t)NT * Ch * 2);
  u16*   ob  = (u16*)w;   w += alg((size_t)NT * Ch * 2);
  u16*   wts = (u16*)w;   w += alg(LAYER_W * Ll * 2);

  // ---- weight prep: f32 -> bf16, transposed to [N,K] for B fragments ----
  for (int l = 0; l < Ll; ++l) {
    u16* base = wts + (size_t)l * LAYER_W;
    dim3 b(256);
    transpose_bf16<<<dim3(64), b, 0, stream>>>(LP(l, I_gine_w1), base + 0,      Ch, Ch);
    transpose_bf16<<<dim3(64), b, 0, stream>>>(LP(l, I_gine_w2), base + 16384,  Ch, Ch);
    transpose_bf16<<<dim3(64), b, 0, stream>>>(LP(l, I_wq),      base + 32768,  Ch, Ch);
    transpose_bf16<<<dim3(64), b, 0, stream>>>(LP(l, I_wk),      base + 49152,  Ch, Ch);
    transpose_bf16<<<dim3(64), b, 0, stream>>>(LP(l, I_wv),      base + 65536,  Ch, Ch);
    transpose_bf16<<<dim3(64), b, 0, stream>>>(LP(l, I_wo),      base + 81920,  Ch, Ch);
    transpose_bf16<<<dim3(128), b, 0, stream>>>(LP(l, I_ffn_w1), base + 98304,  Ch, 2 * Ch);
    transpose_bf16<<<dim3(128), b, 0, stream>>>(LP(l, I_ffn_w2), base + 131072, 2 * Ch, Ch);
  }

  // ---- encoders ----
  encode_kernel<<<dim3(NT), dim3(Ch), 0, stream>>>(ep, x, pe, h, hb);
  bond_kernel<<<dim3(Ee), dim3(Ch), 0, stream>>>(bp, ea, e);

  const int* src = ei;
  const int* dst = ei + Ee;
  const int NC = NT * Ch;

  for (int l = 0; l < Ll; ++l) {
    u16* base = wts + (size_t)l * LAYER_W;
    // --- GINE local message passing ---
    zero_f32<<<dim3(NC / 256), dim3(256), 0, stream>>>(agg, NC);
    gine_msg<<<dim3((Ee * Ch) / 256), dim3(256), 0, stream>>>(h, e, src, dst, agg);
    gine_combine<<<dim3(NC / 256), dim3(256), 0, stream>>>(h, agg, LP(l, I_eps), xb);
    gemm_launch(true,  xb,  base + 0,     LP(l, I_gine_b1), nullptr, nullptr, xb2, NT, Ch, Ch, stream);
    gemm_launch(false, xb2, base + 16384, LP(l, I_gine_b2), h, t0, nullptr, NT, Ch, Ch, stream);
    bn_kernel<<<dim3(Ch), dim3(256), 0, stream>>>(t0, LP(l, I_bn1_g), LP(l, I_bn1_b), t0, nullptr, NT);
    // --- hop-decayed attention ---
    gemm_launch(false, hb, base + 32768, LP(l, I_bq), nullptr, nullptr, qb, NT, Ch, Ch, stream);
    gemm_launch(false, hb, base + 49152, LP(l, I_bk), nullptr, nullptr, kb, NT, Ch, Ch, stream);
    gemm_launch(false, hb, base + 65536, LP(l, I_bv), nullptr, nullptr, vb, NT, Ch, Ch, stream);
    attn_kernel<<<dim3(Bg * Hh * 16), dim3(32), 0, stream>>>(qb, kb, vb, sph, hop, ob);
    gemm_launch(false, ob, base + 81920, LP(l, I_bo), h, t1, nullptr, NT, Ch, Ch, stream);
    bn_kernel<<<dim3(Ch), dim3(256), 0, stream>>>(t1, LP(l, I_bn2_g), LP(l, I_bn2_b), t1, nullptr, NT);
    // --- combine + FFN ---
    ew_add<<<dim3(NC / 256), dim3(256), 0, stream>>>(t0, t1, t2, xb);
    gemm_launch(true,  xb,  base + 98304,  LP(l, I_ffn_b1), nullptr, nullptr, xb2, NT, Ch, 2 * Ch, stream);
    gemm_launch(false, xb2, base + 131072, LP(l, I_ffn_b2), t2, t0, nullptr, NT, 2 * Ch, Ch, stream);
    bn_kernel<<<dim3(Ch), dim3(256), 0, stream>>>(t0, LP(l, I_bn3_g), LP(l, I_bn3_b), h, hb, NT);
  }

  pool_kernel<<<dim3((Bg * Ch) / 256), dim3(256), 0, stream>>>(h, (float*)d_out);
}
